// NetVLAD_13280038880026
// MI455X (gfx1250) — matrix operand order repaired and checked by reading, not compiled, not run on hardware
//
#include <hip/hip_runtime.h>
#include <math.h>

// ---------------------------------------------------------------------------
// NetVLAD + CNN pipeline for MI455X (gfx1250, wave32, WMMA).
// Heavy GEMMs (convs as implicit GEMM, VLAD einsum, MLP) use
// v_wmma_f32_16x16x32_f16 with split-f16 (hi/lo) error compensation:
//   fp32 value v = hi + lo (both f16);  A*B ~= Ah*Bh + Ah*Bl + Al*Bh
// accumulated in fp32 -> ~fp32 accuracy at f16 matrix-pipe rate.
//
// Fragment loads are vectorized: per the ISA VGPR layouts, an A-fragment
// lane holds K in {8g..8g+7} u {16+8g..23+8g} (two contiguous 8-half runs)
// and a B-fragment lane holds the same K set (two 8-half runs), so
// K-contiguous operands are loaded straight from global as v8h/v16h
// (global_load_b128) and the conv im2col tile is staged in LDS in
// fragment order (ds_load_b128 reads).
// ---------------------------------------------------------------------------

typedef _Float16 h16;
typedef __attribute__((ext_vector_type(16))) _Float16 v16h;
typedef __attribute__((ext_vector_type(8)))  _Float16 v8h;
typedef __attribute__((ext_vector_type(8)))  float    v8f;

#define WMMA_F16(A, B, C) \
  __builtin_amdgcn_wmma_f32_16x16x32_f16(false, (A), false, (B), (short)0, (C), false, false)

static __device__ __forceinline__ v16h cat16(v8h a, v8h b) {
  return __builtin_shufflevector(a, b, 0, 1, 2, 3, 4, 5, 6, 7,
                                       8, 9, 10, 11, 12, 13, 14, 15);
}

// ---------------------------------------------------------------------------
// Elementwise split of fp32 into f16 hi + f16 lo residual.
// ---------------------------------------------------------------------------
__global__ void split_hilo(const float* __restrict__ in, h16* __restrict__ hi,
                           h16* __restrict__ lo, int n) {
  int i = blockIdx.x * blockDim.x + threadIdx.x;
  if (i >= n) return;
  float v = in[i];
  h16 h = (h16)v;
  hi[i] = h;
  lo[i] = (h16)(v - (float)h);
}

// ---------------------------------------------------------------------------
// Conv-weight split with tap padding 25 -> 32 (zeros), so A fragments are
// two aligned contiguous 8-half runs per lane.  in: [rows][25] fp32,
// out: [rows][32] f16 hi/lo, rows = OC*Cin.
// ---------------------------------------------------------------------------
__global__ void split_pad_w(const float* __restrict__ in, h16* __restrict__ hi,
                            h16* __restrict__ lo, int total /* rows*32 */) {
  int i = blockIdx.x * blockDim.x + threadIdx.x;
  if (i >= total) return;
  int t = i & 31;
  int r = i >> 5;
  float v = (t < 25) ? in[(size_t)r * 25 + t] : 0.f;
  h16 h = (h16)v;
  hi[i] = h;
  lo[i] = (h16)(v - (float)h);
}

// ---------------------------------------------------------------------------
// Soft-assignment: sa[n,k,p] = conv_w[k,:]·x[n,:,p] + b[k]; softmax over k;
// store a as f16 hi/lo for the WMMA einsum; accumulate asum[n,k] = sum_p a.
// ---------------------------------------------------------------------------
__global__ void softassign(const float* __restrict__ x, const float* __restrict__ w,
                           const float* __restrict__ b, h16* __restrict__ a_hi,
                           h16* __restrict__ a_lo, float* __restrict__ asum) {
  __shared__ float wl[64 * 128];
  __shared__ float bl[64];
  int tid = threadIdx.x;
  for (int i = tid; i < 64 * 128; i += 256) wl[i] = w[i];
  if (tid < 64) bl[tid] = b[tid];
  __syncthreads();

  int n = blockIdx.y;
  int p = blockIdx.x * 256 + tid;
  bool valid = (p < 8000);

  float sa[64];
  for (int k = 0; k < 64; ++k) sa[k] = bl[k];
  if (valid) {
    const float* xb = x + (size_t)n * 128 * 8000 + p;
    for (int c = 0; c < 128; ++c) {
      float xv = xb[(size_t)c * 8000];
      for (int k = 0; k < 64; ++k) sa[k] = fmaf(wl[k * 128 + c], xv, sa[k]);
    }
  }
  float mx = sa[0];
  for (int k = 1; k < 64; ++k) mx = fmaxf(mx, sa[k]);
  float s = 0.f;
  for (int k = 0; k < 64; ++k) { sa[k] = expf(sa[k] - mx); s += sa[k]; }
  float inv = 1.f / s;

  int ln = tid & 31;
  for (int k = 0; k < 64; ++k) {
    float a = valid ? sa[k] * inv : 0.f;
    if (valid) {
      h16 h = (h16)a;
      size_t idx = ((size_t)n * 64 + k) * 8000 + p;
      a_hi[idx] = h;
      a_lo[idx] = (h16)(a - (float)h);
    }
    float r = a;                      // wave32 reduction, lane0 -> atomic
    for (int off = 16; off > 0; off >>= 1) r += __shfl_down(r, off);
    if (ln == 0) atomicAdd(&asum[n * 64 + k], r);
  }
}

// ---------------------------------------------------------------------------
// VLAD einsum: vlad[n,k,c] = sum_p a[n,k,p] * x[n,c,p].
// K-dim = p is contiguous for both operands -> all fragments load straight
// from global as b128 vectors; no LDS.  4 waves/block, wave owns a ktile.
// ---------------------------------------------------------------------------
__global__ void vlad_wmma(const h16* __restrict__ a_hi, const h16* __restrict__ a_lo,
                          const h16* __restrict__ x_hi, const h16* __restrict__ x_lo,
                          float* __restrict__ vacc) {
  int n = blockIdx.y, c0 = blockIdx.x * 16;
  int wv = threadIdx.x >> 5, ln = threadIdx.x & 31;
  int m = ln & 15, g = ln >> 4;

  const h16* abh = a_hi + ((size_t)n * 64 + wv * 16 + m) * 8000 + 8 * g;
  const h16* abl = a_lo + ((size_t)n * 64 + wv * 16 + m) * 8000 + 8 * g;
  const h16* xbh = x_hi + ((size_t)n * 128 + c0 + m) * 8000 + 8 * g;
  const h16* xbl = x_lo + ((size_t)n * 128 + c0 + m) * 8000 + 8 * g;

  v8f acc = {};
  for (int p0 = 0; p0 < 8000; p0 += 32) {
    v16h ah = cat16(*(const v8h*)(abh + p0), *(const v8h*)(abh + p0 + 16));
    v16h al = cat16(*(const v8h*)(abl + p0), *(const v8h*)(abl + p0 + 16));
    v16h bh = cat16(*(const v8h*)(xbh + p0), *(const v8h*)(xbh + p0 + 16));
    v16h bl = cat16(*(const v8h*)(xbl + p0), *(const v8h*)(xbl + p0 + 16));
    acc = WMMA_F16(ah, bh, acc);
    acc = WMMA_F16(ah, bl, acc);
    acc = WMMA_F16(al, bh, acc);
  }
  for (int v = 0; v < 8; ++v) {
    int k = wv * 16 + v + 8 * g;
    vacc[((size_t)n * 64 + k) * 128 + c0 + m] = acc[v];
  }
}

// ---------------------------------------------------------------------------
// vlad residual + intra L2-norm per (n,k) row of 128; accumulate per-n
// global sumsq for the outer norm.
// ---------------------------------------------------------------------------
__global__ void vlad_norm(const float* __restrict__ vacc, const float* __restrict__ asum,
                          const float* __restrict__ centers, float* __restrict__ feat,
                          float* __restrict__ gss) {
  __shared__ float red[128];
  int n = blockIdx.y, k = blockIdx.x, c = threadIdx.x;
  float v = vacc[((size_t)n * 64 + k) * 128 + c] - asum[n * 64 + k] * centers[k * 128 + c];
  red[c] = v * v;
  __syncthreads();
  for (int s = 64; s > 0; s >>= 1) { if (c < s) red[c] += red[c + s]; __syncthreads(); }
  float ss = red[0];
  float scale = 1.f / fmaxf(sqrtf(ss), 1e-12f);
  feat[(size_t)n * 9216 + k * 128 + c] = v * scale;
  if (c == 0) atomicAdd(&gss[n], ss * scale * scale);
}

__global__ void vlad_gscale(float* __restrict__ feat, const float* __restrict__ gss,
                            int total) {
  int i = blockIdx.x * blockDim.x + threadIdx.x;
  if (i >= total) return;
  int n = i >> 13;                                  // / 8192
  float scale = 1.f / fmaxf(sqrtf(gss[n]), 1e-12f);
  feat[(size_t)n * 9216 + (i & 8191)] *= scale;
}

// ---------------------------------------------------------------------------
// 5x5 conv (pad 2) as implicit GEMM.  K-dim = Cin chunks of 32 (25 taps +
// 7 zero taps, pre-padded in the weights).  A fragments: direct global
// b128 loads from padded weights.  B (im2col) tile: staged in LDS in
// fragment order -> each lane reads its v16h as two ds_load_b128.
// Block = 4 waves; wave w owns outC tile oc0 + 16w; B tile shared.
// ---------------------------------------------------------------------------
__global__ void conv5x5_wmma(const h16* __restrict__ in_hi, const h16* __restrict__ in_lo,
                             const h16* __restrict__ w_hi, const h16* __restrict__ w_lo,
                             const float* __restrict__ bias, float* __restrict__ out,
                             int Cin, int H, int W, int OC, int leaky) {
  __shared__ v16h BhF[32], BlF[32];     // fragment-order: lane l -> BhF[l]
  int P = H * W;
  int n = blockIdx.z;
  int oc0 = blockIdx.y * 64;
  int p0 = blockIdx.x * 16;
  int tid = threadIdx.x, wv = tid >> 5, ln = tid & 31;
  int m = ln & 15, g = ln >> 4;

  const h16* wbase_h = w_hi + (size_t)(oc0 + wv * 16 + m) * Cin * 32 + 8 * g;
  const h16* wbase_l = w_lo + (size_t)(oc0 + wv * 16 + m) * Cin * 32 + 8 * g;
  h16* bph = (h16*)BhF;
  h16* bpl = (h16*)BlF;

  v8f acc = {};
  for (int c = 0; c < Cin; ++c) {
    const h16* xbh = in_hi + ((size_t)n * Cin + c) * P;
    const h16* xbl = in_lo + ((size_t)n * Cin + c) * P;
    if (c + 1 < Cin) {                 // next channel -> global_prefetch
      __builtin_prefetch(xbh + P, 0, 0);
      __builtin_prefetch(xbl + P, 0, 0);
    }
    __syncthreads();
    // all 128 threads stage the 32x16 im2col tile (512 elems, 4 each)
    for (int i = 0; i < 4; ++i) {
      int e0 = i * 128 + tid;
      int t = e0 >> 4, px = e0 & 15;   // t = tap (K), px = pixel col
      h16 vh = (h16)0, vl = (h16)0;
      int p = p0 + px;
      if (t < 25 && p < P) {
        int y = p / W, xc = p % W;
        int r = t / 5, s = t % 5;
        int yy = y + r - 2, xx = xc + s - 2;
        if (yy >= 0 && yy < H && xx >= 0 && xx < W) {
          vh = xbh[yy * W + xx];
          vl = xbl[yy * W + xx];
        }
      }
      int di = (((((t >> 3) & 1) << 4) + px) << 4) + ((t >> 4) << 3) + (t & 7);  // frag order
      bph[di] = vh;
      bpl[di] = vl;
    }
    __syncthreads();

    v16h ah = cat16(*(const v8h*)(wbase_h + (size_t)c * 32),
                    *(const v8h*)(wbase_h + (size_t)c * 32 + 16));
    v16h al = cat16(*(const v8h*)(wbase_l + (size_t)c * 32),
                    *(const v8h*)(wbase_l + (size_t)c * 32 + 16));
    v16h bh = BhF[ln];
    v16h bl = BlF[ln];
    acc = WMMA_F16(ah, bh, acc);
    acc = WMMA_F16(ah, bl, acc);
    acc = WMMA_F16(al, bh, acc);
  }

  int p = p0 + m;
  if (p < P) {
    for (int v = 0; v < 8; ++v) {
      int oc = oc0 + wv * 16 + v + 8 * g;
      float val = acc[v] + bias[oc];
      if (leaky) val = (val >= 0.f) ? val : 0.2f * val;
      out[((size_t)n * OC + oc) * P + p] = val;
    }
  }
}

// ---------------------------------------------------------------------------
// MaxPool (ph x pw, stride = window) fused with hi/lo split for the next conv.
// ---------------------------------------------------------------------------
__global__ void pool_split(const float* __restrict__ in, h16* __restrict__ ohi,
                           h16* __restrict__ olo, int H, int W, int ph, int pw,
                           int OH, int OW, int total) {
  int i = blockIdx.x * blockDim.x + threadIdx.x;
  if (i >= total) return;
  int ox = i % OW;
  int t = i / OW;
  int oy = t % OH;
  int nc = t / OH;
  const float* base = in + (size_t)nc * H * W;
  float mv = -3.4e38f;
  for (int r = 0; r < ph; ++r)
    for (int s = 0; s < pw; ++s)
      mv = fmaxf(mv, base[(oy * ph + r) * W + ox * pw + s]);
  h16 h = (h16)mv;
  ohi[i] = h;
  olo[i] = (h16)(mv - (float)h);
}

// ---------------------------------------------------------------------------
// Final CNN stage: maxpool over all 125 pixels (5x25) + per-image L2 norm,
// writing into feat[n, 8192 .. 9215].
// ---------------------------------------------------------------------------
__global__ void pool3_l2(const float* __restrict__ h3, float* __restrict__ feat) {
  __shared__ float red[256];
  int n = blockIdx.x, t = threadIdx.x;
  float vals[4];
  float ss = 0.f;
  for (int j = 0; j < 4; ++j) {
    int oc = t + 256 * j;
    const float* b = h3 + ((size_t)n * 1024 + oc) * 125;
    float mv = -3.4e38f;
    for (int p = 0; p < 125; ++p) mv = fmaxf(mv, b[p]);
    vals[j] = mv;
    ss += mv * mv;
  }
  red[t] = ss;
  __syncthreads();
  for (int s = 128; s > 0; s >>= 1) { if (t < s) red[t] += red[t + s]; __syncthreads(); }
  float scale = 1.f / fmaxf(sqrtf(red[0]), 1e-12f);
  for (int j = 0; j < 4; ++j)
    feat[(size_t)n * 9216 + 8192 + t + 256 * j] = vals[j] * scale;
}

// ---------------------------------------------------------------------------
// MLP head: out_pre[n,j] = feat[n,:]·mlp_w[j,:] + b[j].  One WMMA GEMM,
// M = 16 images (single tile), N = 256 in 16 tiles, K = 9216 in 32-chunks.
// K is contiguous for both operands -> direct global b128 fragment loads.
// ---------------------------------------------------------------------------
__global__ void mlp_wmma(const h16* __restrict__ fhi, const h16* __restrict__ flo,
                         const h16* __restrict__ whi, const h16* __restrict__ wlo,
                         const float* __restrict__ bias, float* __restrict__ outp) {
  int j0 = blockIdx.x * 16;
  int ln = threadIdx.x & 31;
  int m = ln & 15, g = ln >> 4;

  const h16* fbh = fhi + (size_t)m * 9216 + 8 * g;
  const h16* fbl = flo + (size_t)m * 9216 + 8 * g;
  const h16* wbh = whi + (size_t)(j0 + m) * 9216 + 8 * g;
  const h16* wbl = wlo + (size_t)(j0 + m) * 9216 + 8 * g;

  v8f acc = {};
  for (int f0 = 0; f0 < 9216; f0 += 32) {
    v16h ah = cat16(*(const v8h*)(fbh + f0), *(const v8h*)(fbh + f0 + 16));
    v16h al = cat16(*(const v8h*)(fbl + f0), *(const v8h*)(fbl + f0 + 16));
    v16h bh = cat16(*(const v8h*)(wbh + f0), *(const v8h*)(wbh + f0 + 16));
    v16h bl = cat16(*(const v8h*)(wbl + f0), *(const v8h*)(wbl + f0 + 16));
    acc = WMMA_F16(ah, bh, acc);
    acc = WMMA_F16(ah, bl, acc);
    acc = WMMA_F16(al, bh, acc);
  }
  for (int v = 0; v < 8; ++v) {
    int n = v + 8 * g;          // row = image index
    int j = j0 + m;             // col = output feature
    outp[n * 256 + j] = acc[v] + bias[j];
  }
}

__global__ void l2norm_rows(const float* __restrict__ in, float* __restrict__ out) {
  __shared__ float red[256];
  int n = blockIdx.x, j = threadIdx.x;
  float v = in[n * 256 + j];
  red[j] = v * v;
  __syncthreads();
  for (int s = 128; s > 0; s >>= 1) { if (j < s) red[j] += red[j + s]; __syncthreads(); }
  float scale = 1.f / fmaxf(sqrtf(red[0]), 1e-12f);
  out[n * 256 + j] = v * scale;
}

// ---------------------------------------------------------------------------
// Host launcher.  Workspace layout (~333 MB) with region reuse:
//   regionA: h1(131MB) -> h2(65.5MB) -> h3(8.2MB)
//   regionB: x_hi+x_lo(65.5MB) -> h1p_hi+h1p_lo(32.8MB)
// ---------------------------------------------------------------------------
extern "C" void kernel_launch(void* const* d_in, const int* in_sizes, int n_in,
                              void* d_out, int out_size, void* d_ws, size_t ws_size,
                              hipStream_t stream) {
  (void)in_sizes; (void)n_in; (void)out_size; (void)ws_size;
  const float* x    = (const float*)d_in[0];
  const float* cent = (const float*)d_in[1];
  const float* cw   = (const float*)d_in[2];
  const float* cb   = (const float*)d_in[3];
  const float* w1   = (const float*)d_in[4];
  const float* b1   = (const float*)d_in[5];
  const float* w2   = (const float*)d_in[6];
  const float* b2   = (const float*)d_in[7];
  const float* w3   = (const float*)d_in[8];
  const float* b3   = (const float*)d_in[9];
  const float* mw   = (const float*)d_in[10];
  const float* mb   = (const float*)d_in[11];
  float* out = (float*)d_out;

  char* ws = (char*)d_ws;
  size_t off = 0;
  auto alloc = [&](size_t bytes) -> char* {
    char* r = ws + off;
    off = (off + bytes + 255) & ~(size_t)255;
    return r;
  };

  char*  regionA = alloc(131072000);              // h1 / h2 / h3
  char*  regionB = alloc(65536000);               // x hi/lo -> h1p hi/lo
  h16*   a_hi = (h16*)alloc(16384000);
  h16*   a_lo = (h16*)alloc(16384000);
  h16*   w1h  = (h16*)alloc(2097152);             // padded [256][128][32]
  h16*   w1l  = (h16*)alloc(2097152);
  h16*   w2h  = (h16*)alloc(8388608);             // padded [512][256][32]
  h16*   w2l  = (h16*)alloc(8388608);
  h16*   w3h  = (h16*)alloc(33554432);            // padded [1024][512][32]
  h16*   w3l  = (h16*)alloc(33554432);
  h16*   mwh  = (h16*)alloc(4718592);
  h16*   mwl  = (h16*)alloc(4718592);
  h16*   h2ph = (h16*)alloc(2048000);
  h16*   h2pl = (h16*)alloc(2048000);
  float* vacc = (float*)alloc(524288);
  float* feat = (float*)alloc(589824);
  h16*   fh   = (h16*)alloc(294912);
  h16*   fl   = (h16*)alloc(294912);
  float* mpre = (float*)alloc(16384);
  float* asum = (float*)alloc(4096);
  float* gss  = (float*)alloc(64);                // contiguous after asum

  h16*   x_hi = (h16*)regionB;
  h16*   x_lo = (h16*)(regionB + 32768000);
  float* h1   = (float*)regionA;
  h16*   h1ph = (h16*)regionB;                    // reuse after conv1
  h16*   h1pl = (h16*)(regionB + 16384000);
  float* h2   = (float*)regionA;                  // reuse after pool1
  float* h3   = (float*)regionA;                  // reuse after pool2

  hipMemsetAsync(asum, 0, 4096 + 64, stream);     // asum + gss (adjacent)

  // --- precision splits of inputs / weights ---
  split_hilo<<<(16384000 + 255) / 256, 256, 0, stream>>>(x, x_hi, x_lo, 16384000);
  split_pad_w<<<(1048576  + 255) / 256, 256, 0, stream>>>(w1, w1h, w1l, 1048576);
  split_pad_w<<<(4194304  + 255) / 256, 256, 0, stream>>>(w2, w2h, w2l, 4194304);
  split_pad_w<<<(16777216 + 255) / 256, 256, 0, stream>>>(w3, w3h, w3l, 16777216);
  split_hilo<<<(2359296  + 255) / 256, 256, 0, stream>>>(mw, mwh, mwl, 2359296);

  // --- VLAD branch ---
  softassign<<<dim3(32, 16), 256, 0, stream>>>(x, cw, cb, a_hi, a_lo, asum);
  vlad_wmma<<<dim3(8, 16), 128, 0, stream>>>(a_hi, a_lo, x_hi, x_lo, vacc);
  vlad_norm<<<dim3(64, 16), 128, 0, stream>>>(vacc, asum, cent, feat, gss);
  vlad_gscale<<<(131072 + 255) / 256, 256, 0, stream>>>(feat, gss, 131072);

  // --- CNN branch ---
  conv5x5_wmma<<<dim3(500, 4, 16), 128, 0, stream>>>(x_hi, x_lo, w1h, w1l, b1, h1,
                                                     128, 40, 200, 256, 1);
  pool_split<<<(8192000 + 255) / 256, 256, 0, stream>>>(h1, h1ph, h1pl,
                                                        40, 200, 2, 2, 20, 100, 8192000);
  conv5x5_wmma<<<dim3(125, 8, 16), 128, 0, stream>>>(h1ph, h1pl, w2h, w2l, b2, h2,
                                                     256, 20, 100, 512, 1);
  pool_split<<<(1024000 + 255) / 256, 256, 0, stream>>>(h2, h2ph, h2pl,
                                                        20, 100, 4, 4, 5, 25, 1024000);
  conv5x5_wmma<<<dim3(8, 16, 16), 128, 0, stream>>>(h2ph, h2pl, w3h, w3l, b3, h3,
                                                    512, 5, 25, 1024, 0);
  pool3_l2<<<16, 256, 0, stream>>>(h3, feat);

  // --- head ---
  split_hilo<<<(147456 + 255) / 256, 256, 0, stream>>>(feat, fh, fl, 147456);
  mlp_wmma<<<16, 32, 0, stream>>>(fh, fl, mwh, mwl, mb, mpre);
  l2norm_rows<<<16, 256, 0, stream>>>(mpre, out);
}
